// GNNBranch_76587856822781
// MI455X (gfx1250) — matrix operand order repaired and checked by reading, not compiled, not run on hardware
//
#include <hip/hip_runtime.h>
#include <hip/hip_bf16.h>

// ---------------------------------------------------------------------------
// CDNA5 (gfx1250) GNN pipeline: WMMA bf16 GEMMs + fused gather/scatter edges.
// ---------------------------------------------------------------------------

typedef __attribute__((ext_vector_type(16))) __bf16 v16bf;
typedef __attribute__((ext_vector_type(8)))  float  v8f;

#define ACT_NONE 0
#define ACT_RELU 1
#define ACT_TANH 2

__device__ __forceinline__ unsigned short f2bf(float f) {
  // round-to-nearest-even f32 -> bf16
  unsigned u = __float_as_uint(f);
  u += 0x7FFFu + ((u >> 16) & 1u);
  return (unsigned short)(u >> 16);
}

__device__ __forceinline__ unsigned pack2(unsigned short lo, unsigned short hi) {
  return (unsigned)lo | ((unsigned)hi << 16);
}

// ---------------------------------------------------------------------------
// Generic GEMM: C[M,Ncols] = act(A[M,K] @ W[K,Ncols] + bias)
// Block: 256 threads (8 waves). Block tile 64(M) x 64(N).
// Each wave computes one 16x32 slab: 2 WMMAs per K-step sharing the A frag.
// A/B staged in LDS pre-swizzled into the ISA bf16 fragment layouts so each
// lane reads its 16-element fragment contiguously (2 x ds_load_b128).
// ---------------------------------------------------------------------------
__global__ __launch_bounds__(256) void gemm_wmma_kernel(
    const float* __restrict__ A, const float* __restrict__ W,
    const float* __restrict__ bias, float* __restrict__ C,
    int M, int K, int Ncols, int act)
{
  __shared__ __align__(16) unsigned short lds_a[4 * 512];  // 64(m) x 32(k)
  __shared__ __align__(16) unsigned short lds_b[4 * 512];  // 32(k) x 64(n)

  const int tid  = threadIdx.x;
  const int lane = tid & 31;
  const int wave = tid >> 5;
  const int mt   = wave >> 1;        // 0..3  (m tile)
  const int ng   = wave & 1;         // 0..1  (n tile pair: 2*ng, 2*ng+1)
  const int m0   = blockIdx.x * 64;
  const int n0   = blockIdx.y * 64;

  // ---- A fill mapping: thread owns row am, 8 consecutive k (akq..akq+7) ----
  const int am   = tid >> 2;          // 0..63
  const int akq  = (tid & 3) * 8;     // 0,8,16,24
  const int amt  = am >> 4;
  const int aln  = (am & 15) + ((akq & 8) ? 16 : 0);
  unsigned short* const asl = &lds_a[amt * 512 + aln * 16 + ((akq & 16) ? 8 : 0)];
  const int agrow = m0 + am;
  const float* const arow = A + (long)agrow * K;

  // ---- B fill mapping: thread owns col bnl, 8 consecutive k (bkq..bkq+7) ----
  const int bnl  = tid & 63;          // 0..63
  const int bkq  = (tid >> 6) * 8;    // 0,8,16,24
  const int bnt  = bnl >> 4;
  const int bln  = (bnl & 15) + ((bkq & 8) ? 16 : 0);
  unsigned short* const bsl = &lds_b[bnt * 512 + bln * 16 + ((bkq & 16) ? 8 : 0)];
  const float* const bcol = W + (long)n0 + bnl;

  v8f acc0 = {};
  v8f acc1 = {};

  for (int k0 = 0; k0 < K; k0 += 32) {
    // --- A tile fill: fast path = 2x global_load_b128, 1x ds_store_b128 ---
    {
      unsigned short t0, t1, t2, t3, t4, t5, t6, t7;
      if (agrow < M && (k0 + akq + 8) <= K) {
        const float4 f0 = *reinterpret_cast<const float4*>(arow + k0 + akq);
        const float4 f1 = *reinterpret_cast<const float4*>(arow + k0 + akq + 4);
        t0 = f2bf(f0.x); t1 = f2bf(f0.y); t2 = f2bf(f0.z); t3 = f2bf(f0.w);
        t4 = f2bf(f1.x); t5 = f2bf(f1.y); t6 = f2bf(f1.z); t7 = f2bf(f1.w);
      } else {
        unsigned short tt[8];
#pragma unroll
        for (int i = 0; i < 8; ++i) {
          const int gk = k0 + akq + i;
          tt[i] = (agrow < M && gk < K) ? f2bf(arow[gk]) : (unsigned short)0;
        }
        t0 = tt[0]; t1 = tt[1]; t2 = tt[2]; t3 = tt[3];
        t4 = tt[4]; t5 = tt[5]; t6 = tt[6]; t7 = tt[7];
      }
      uint4 u;
      u.x = pack2(t0, t1); u.y = pack2(t2, t3);
      u.z = pack2(t4, t5); u.w = pack2(t6, t7);
      *reinterpret_cast<uint4*>(asl) = u;
      if (k0 + 32 < K) __builtin_prefetch(arow + k0 + 32 + akq, 0, 1);
    }
    // --- B tile fill: 8 coalesced b32 loads (stride Ncols), 1x ds_store_b128 ---
    {
      unsigned short tt[8];
      const float* wp = bcol + (long)(k0 + bkq) * Ncols;
      if (k0 + bkq + 8 <= K) {
#pragma unroll
        for (int i = 0; i < 8; ++i) tt[i] = f2bf(wp[(long)i * Ncols]);
      } else {
#pragma unroll
        for (int i = 0; i < 8; ++i) {
          const int gk = k0 + bkq + i;
          tt[i] = (gk < K) ? f2bf(wp[(long)i * Ncols]) : (unsigned short)0;
        }
      }
      uint4 u;
      u.x = pack2(tt[0], tt[1]); u.y = pack2(tt[2], tt[3]);
      u.z = pack2(tt[4], tt[5]); u.w = pack2(tt[6], tt[7]);
      *reinterpret_cast<uint4*>(bsl) = u;
    }
    __syncthreads();
    v16bf a  = *reinterpret_cast<const v16bf*>(&lds_a[mt * 512 + lane * 16]);
    v16bf b0 = *reinterpret_cast<const v16bf*>(&lds_b[(2 * ng) * 512 + lane * 16]);
    v16bf b1 = *reinterpret_cast<const v16bf*>(&lds_b[(2 * ng + 1) * 512 + lane * 16]);
    acc0 = __builtin_amdgcn_wmma_f32_16x16x32_bf16(false, a, false, b0,
                                                   (short)0, acc0, false, false);
    acc1 = __builtin_amdgcn_wmma_f32_16x16x32_bf16(false, a, false, b1,
                                                   (short)0, acc1, false, false);
    __syncthreads();
  }

  // epilogue: C frag -> VGPR v: M=v (lanes 0-15) / M=v+8 (lanes 16-31), N=lane&15
  const int col0  = n0 + (2 * ng) * 16 + (lane & 15);
  const float bv0 = bias ? bias[col0] : 0.0f;
  const float bv1 = bias ? bias[col0 + 16] : 0.0f;
  const int rbase = m0 + mt * 16 + ((lane & 16) ? 8 : 0);
#pragma unroll
  for (int v = 0; v < 8; ++v) {
    const int row = rbase + v;
    if (row < M) {
      float o0 = acc0[v] + bv0;
      float o1 = acc1[v] + bv1;
      if (act == ACT_RELU)      { o0 = fmaxf(o0, 0.0f); o1 = fmaxf(o1, 0.0f); }
      else if (act == ACT_TANH) { o0 = tanhf(o0);       o1 = tanhf(o1); }
      C[(long)row * Ncols + col0]      = o0;
      C[(long)row * Ncols + col0 + 16] = o1;
    }
  }
}

// ---------------------------------------------------------------------------
// Fused GINE edge pass: ee = Ebf @ W + b (WMMA, K=128, N=256), then
// msg = relu(h[src] + ee), atomic scatter-add into z[dst].
// Same 64x64 tile structure; A is already bf16 (raw uint4 loads).
// ---------------------------------------------------------------------------
__global__ __launch_bounds__(256) void edge_gemm_scatter_kernel(
    const unsigned short* __restrict__ Ebf, const float* __restrict__ W,
    const float* __restrict__ bias, const float* __restrict__ h,
    float* __restrict__ z, const int* __restrict__ src,
    const int* __restrict__ dst, int E, int K, int Ncols)
{
  __shared__ __align__(16) unsigned short lds_a[4 * 512];
  __shared__ __align__(16) unsigned short lds_b[4 * 512];
  __shared__ int lds_src[64];
  __shared__ int lds_dst[64];

  const int tid  = threadIdx.x;
  const int lane = tid & 31;
  const int wave = tid >> 5;
  const int mt   = wave >> 1;
  const int ng   = wave & 1;
  const int e0   = blockIdx.x * 64;
  const int n0   = blockIdx.y * 64;

  if (tid < 64) {
    const int e = e0 + tid;
    lds_src[tid] = (e < E) ? src[e] : 0;
    lds_dst[tid] = (e < E) ? dst[e] : 0;
  }

  const int am  = tid >> 2;
  const int akq = (tid & 3) * 8;
  const int amt = am >> 4;
  const int aln = (am & 15) + ((akq & 8) ? 16 : 0);
  unsigned short* const asl = &lds_a[amt * 512 + aln * 16 + ((akq & 16) ? 8 : 0)];
  const int age = e0 + am;
  const unsigned short* const arow = Ebf + (long)age * K;

  const int bnl = tid & 63;
  const int bkq = (tid >> 6) * 8;
  const int bnt = bnl >> 4;
  const int bln = (bnl & 15) + ((bkq & 8) ? 16 : 0);
  unsigned short* const bsl = &lds_b[bnt * 512 + bln * 16 + ((bkq & 16) ? 8 : 0)];
  const float* const bcol = W + (long)n0 + bnl;

  v8f acc0 = {};
  v8f acc1 = {};

  for (int k0 = 0; k0 < K; k0 += 32) {
    // A: 8 consecutive bf16 = one uint4 load, one uint4 LDS store
    {
      uint4 u = {0u, 0u, 0u, 0u};
      if (age < E) u = *reinterpret_cast<const uint4*>(arow + k0 + akq);
      *reinterpret_cast<uint4*>(asl) = u;
      if (k0 + 32 < K) __builtin_prefetch(arow + k0 + 32 + akq, 0, 1);
    }
    // B: 8 coalesced b32 loads, convert, one uint4 LDS store (K multiple of 32)
    {
      const float* wp = bcol + (long)(k0 + bkq) * Ncols;
      unsigned short tt[8];
#pragma unroll
      for (int i = 0; i < 8; ++i) tt[i] = f2bf(wp[(long)i * Ncols]);
      uint4 u;
      u.x = pack2(tt[0], tt[1]); u.y = pack2(tt[2], tt[3]);
      u.z = pack2(tt[4], tt[5]); u.w = pack2(tt[6], tt[7]);
      *reinterpret_cast<uint4*>(bsl) = u;
    }
    __syncthreads();
    v16bf a  = *reinterpret_cast<const v16bf*>(&lds_a[mt * 512 + lane * 16]);
    v16bf b0 = *reinterpret_cast<const v16bf*>(&lds_b[(2 * ng) * 512 + lane * 16]);
    v16bf b1 = *reinterpret_cast<const v16bf*>(&lds_b[(2 * ng + 1) * 512 + lane * 16]);
    acc0 = __builtin_amdgcn_wmma_f32_16x16x32_bf16(false, a, false, b0,
                                                   (short)0, acc0, false, false);
    acc1 = __builtin_amdgcn_wmma_f32_16x16x32_bf16(false, a, false, b1,
                                                   (short)0, acc1, false, false);
    __syncthreads();
  }

  const int col0  = n0 + (2 * ng) * 16 + (lane & 15);
  const float bv0 = bias[col0];
  const float bv1 = bias[col0 + 16];
  const int rb    = mt * 16 + ((lane & 16) ? 8 : 0);
#pragma unroll
  for (int v = 0; v < 8; ++v) {
    const int el = rb + v;
    const int e  = e0 + el;
    if (e < E) {
      const int s = lds_src[el];
      const int d = lds_dst[el];
      const long hs = (long)s * Ncols;
      const long zd = (long)d * Ncols;
      float o0 = fmaxf(acc0[v] + bv0 + h[hs + col0], 0.0f);
      float o1 = fmaxf(acc1[v] + bv1 + h[hs + col0 + 16], 0.0f);
      atomicAdd(&z[zd + col0], o0);
      atomicAdd(&z[zd + col0 + 16], o1);
    }
  }
}

// ---------------------------------------------------------------------------
// Small helper kernels
// ---------------------------------------------------------------------------
__global__ void fill_zero_kernel(float* __restrict__ p, long n) {
  long i = (long)blockIdx.x * blockDim.x + threadIdx.x;
  const long stride = (long)gridDim.x * blockDim.x;
  for (; i < n; i += stride) p[i] = 0.0f;
}

// e = edge_attr[E,6] @ edge_w[6,128] + edge_b, stored as bf16
__global__ __launch_bounds__(128) void edge_embed_kernel(
    const float* __restrict__ ea, const float* __restrict__ ew,
    const float* __restrict__ eb, unsigned short* __restrict__ ebf, int E)
{
  const int e = blockIdx.x;
  const int c = threadIdx.x;
  if (e >= E) return;
  float acc = eb[c];
#pragma unroll
  for (int j = 0; j < 6; ++j) acc += ea[(long)e * 6 + j] * ew[j * 128 + c];
  ebf[(long)e * 128 + c] = f2bf(acc);
}

__global__ void scale_copy_kernel(const float* __restrict__ h, float* __restrict__ z,
                                  const float* __restrict__ eps, int layer, long n) {
  const float s = 1.0f + eps[layer];
  long i = (long)blockIdx.x * blockDim.x + threadIdx.x;
  const long stride = (long)gridDim.x * blockDim.x;
  for (; i < n; i += stride) z[i] = s * h[i];
}

__global__ void count_nodes_kernel(const int* __restrict__ batch, float* __restrict__ cnt, int N) {
  const int i = blockIdx.x * blockDim.x + threadIdx.x;
  if (i < N) atomicAdd(&cnt[batch[i]], 1.0f);
}

// BatchNorm statistics: per-channel sum / sumsq (channel = threadIdx.x, 256 wide)
__global__ __launch_bounds__(256) void bn_stats_kernel(
    const float* __restrict__ z, float* __restrict__ bnsum, float* __restrict__ bnss,
    int N, int rpb)
{
  const int c  = threadIdx.x;
  const int r0 = blockIdx.x * rpb;
  const int r1 = min(r0 + rpb, N);
  float s = 0.0f, ss = 0.0f;
  for (int r = r0; r < r1; ++r) {
    const float v = z[(long)r * 256 + c];
    s += v; ss += v * v;
  }
  atomicAdd(&bnsum[c], s);
  atomicAdd(&bnss[c], ss);
}

__global__ __launch_bounds__(256) void bn_finalize_kernel(
    const float* __restrict__ bnsum, const float* __restrict__ bnss,
    const float* __restrict__ gamma, const float* __restrict__ beta,
    float* __restrict__ scale, float* __restrict__ shift, int N)
{
  const int c = threadIdx.x;
  const float mu  = bnsum[c] / (float)N;
  const float var = bnss[c] / (float)N - mu * mu;
  const float rs  = rsqrtf(var + 1e-5f);
  const float sc  = gamma[c] * rs;
  scale[c] = sc;
  shift[c] = beta[c] - mu * sc;
}

// Apply BN affine; accumulate per-graph channel sums (batch is sorted -> run-length)
__global__ __launch_bounds__(256) void bn_apply_gsum_kernel(
    float* __restrict__ z, const int* __restrict__ batch, float* __restrict__ gsum,
    const float* __restrict__ scale, const float* __restrict__ shift, int N, int rpb)
{
  const int c  = threadIdx.x;
  const int r0 = blockIdx.x * rpb;
  const int r1 = min(r0 + rpb, N);
  const float sc = scale[c], sh = shift[c];
  int gcur = -1; float acc = 0.0f;
  for (int r = r0; r < r1; ++r) {
    const float v = z[(long)r * 256 + c] * sc + sh;
    z[(long)r * 256 + c] = v;
    const int g = batch[r];
    if (g != gcur) {
      if (gcur >= 0) atomicAdd(&gsum[(long)gcur * 256 + c], acc);
      gcur = g; acc = 0.0f;
    }
    acc += v;
  }
  if (gcur >= 0) atomicAdd(&gsum[(long)gcur * 256 + c], acc);
}

// PairNorm center: z -= gmean[batch]; accumulate per-graph sum of squares
__global__ __launch_bounds__(256) void pn_center_kernel(
    float* __restrict__ z, const int* __restrict__ batch,
    const float* __restrict__ gsum, const float* __restrict__ cnt,
    float* __restrict__ msq, int N, int rpb)
{
  const int c  = threadIdx.x;
  const int r0 = blockIdx.x * rpb;
  const int r1 = min(r0 + rpb, N);
  int gcur = -1; float acc = 0.0f; float gm = 0.0f;
  for (int r = r0; r < r1; ++r) {
    const int g = batch[r];
    if (g != gcur) {
      if (gcur >= 0) atomicAdd(&msq[gcur], acc);
      gcur = g; acc = 0.0f;
      gm = gsum[(long)g * 256 + c] / fmaxf(cnt[g], 1.0f);
    }
    const float v = z[(long)r * 256 + c] - gm;
    z[(long)r * 256 + c] = v;
    acc += v * v;
  }
  if (gcur >= 0) atomicAdd(&msq[gcur], acc);
}

// PairNorm scale + relu into h
__global__ __launch_bounds__(256) void pn_scale_relu_kernel(
    const float* __restrict__ z, const int* __restrict__ batch,
    const float* __restrict__ msq, const float* __restrict__ cnt,
    float* __restrict__ h, int N, int rpb)
{
  const int c  = threadIdx.x;
  const int r0 = blockIdx.x * rpb;
  const int r1 = min(r0 + rpb, N);
  int gcur = -1; float s = 1.0f;
  for (int r = r0; r < r1; ++r) {
    const int g = batch[r];
    if (g != gcur) {
      gcur = g;
      s = rsqrtf(1e-5f + msq[g] / fmaxf(cnt[g], 1.0f));
    }
    h[(long)r * 256 + c] = fmaxf(z[(long)r * 256 + c] * s, 0.0f);
  }
}

// attention scores: score = t1[r,:128] . w2 + b2 ; w = exp(score); wsum[g] += w
__global__ void att_score_kernel(
    const float* __restrict__ t1, const float* __restrict__ w2,
    const float* __restrict__ b2, const int* __restrict__ batch,
    float* __restrict__ wbuf, float* __restrict__ wsum, int N)
{
  const int r = blockIdx.x * blockDim.x + threadIdx.x;
  if (r >= N) return;
  float s = b2[0];
  const float* tp = t1 + (long)r * 128;
#pragma unroll 8
  for (int j = 0; j < 128; ++j) s += tp[j] * w2[j];
  const float w = expf(s);
  wbuf[r] = w;
  atomicAdd(&wsum[batch[r]], w);
}

// mean / max / attention pooling accumulation (h >= 0 post-relu -> uint max OK)
__global__ __launch_bounds__(256) void pool_accum_kernel(
    const float* __restrict__ h, const int* __restrict__ batch,
    const float* __restrict__ wbuf, const float* __restrict__ wsum,
    float* __restrict__ meanacc, float* __restrict__ maxacc,
    float* __restrict__ attacc, int N, int rpb)
{
  const int c  = threadIdx.x;
  const int r0 = blockIdx.x * rpb;
  const int r1 = min(r0 + rpb, N);
  for (int r = r0; r < r1; ++r) {
    const int g   = batch[r];
    const float v = h[(long)r * 256 + c];
    atomicAdd(&meanacc[(long)g * 256 + c], v);
    atomicMax((unsigned int*)&maxacc[(long)g * 256 + c], __float_as_uint(v));
    const float wn = wbuf[r] / (wsum[g] + 1e-8f);
    atomicAdd(&attacc[(long)g * 256 + c], v * wn);
  }
}

__global__ void final_out_kernel(
    const float* __restrict__ meanacc, const float* __restrict__ maxacc,
    const float* __restrict__ attacc, const float* __restrict__ cnt,
    float* __restrict__ out, int G)
{
  const long idx = (long)blockIdx.x * blockDim.x + threadIdx.x;
  if (idx >= (long)G * 768) return;
  const int g = (int)(idx / 768);
  const int c = (int)(idx % 768);
  float r;
  if (c < 256)      r = meanacc[(long)g * 256 + c] / fmaxf(cnt[g], 1.0f);
  else if (c < 512) r = maxacc[(long)g * 256 + (c - 256)];
  else              r = attacc[(long)g * 256 + (c - 512)];
  out[idx] = r;
}

// ---------------------------------------------------------------------------
// Host orchestration
// ---------------------------------------------------------------------------
static inline void fill0(float* p, long n, hipStream_t s) {
  long blocks = (n + 255) / 256;
  if (blocks > 4096) blocks = 4096;
  if (blocks < 1) blocks = 1;
  fill_zero_kernel<<<(unsigned)blocks, 256, 0, s>>>(p, n);
}

extern "C" void kernel_launch(void* const* d_in, const int* in_sizes, int n_in,
                              void* d_out, int out_size, void* d_ws, size_t ws_size,
                              hipStream_t stream) {
  const float* x        = (const float*)d_in[0];
  const float* edge_at  = (const float*)d_in[1];
  const float* node_w   = (const float*)d_in[2];
  const float* node_b   = (const float*)d_in[3];
  const float* edge_w   = (const float*)d_in[4];
  const float* edge_b   = (const float*)d_in[5];
  const float* cew      = (const float*)d_in[6];   // [L,128,256]
  const float* ceb      = (const float*)d_in[7];   // [L,256]
  const float* mw1      = (const float*)d_in[8];   // [L,256,256]
  const float* mb1      = (const float*)d_in[9];
  const float* mw2      = (const float*)d_in[10];
  const float* mb2      = (const float*)d_in[11];
  const float* eps      = (const float*)d_in[12];  // [L]
  const float* bng      = (const float*)d_in[13];  // [L,256]
  const float* bnb      = (const float*)d_in[14];
  const float* att_w1   = (const float*)d_in[15];  // [256,128]
  const float* att_b1   = (const float*)d_in[16];
  const float* att_w2   = (const float*)d_in[17];  // [128]
  const float* att_b2   = (const float*)d_in[18];
  const int*   eidx     = (const int*)d_in[19];    // [2,E]
  const int*   batch    = (const int*)d_in[20];    // [N]
  float*       out      = (float*)d_out;

  const int N = in_sizes[0] / 48;
  const int E = in_sizes[1] / 6;
  const int L = in_sizes[6] / (128 * 256);
  const int G = out_size / 768;
  const int HID = 256;

  // workspace carve-out
  char* ws = (char*)d_ws;
  size_t off = 0;
  auto carve = [&](size_t bytes) -> void* {
    void* p = ws + off;
    off += (bytes + 255) & ~(size_t)255;
    return p;
  };
  float* h    = (float*)carve((size_t)N * HID * 4);
  float* z    = (float*)carve((size_t)N * HID * 4);
  float* t    = (float*)carve((size_t)N * HID * 4);
  unsigned short* ebf = (unsigned short*)carve((size_t)E * 128 * 2);
  float* gsum = (float*)carve((size_t)G * HID * 4);
  float* maxb = (float*)carve((size_t)G * HID * 4);
  float* atta = (float*)carve((size_t)G * HID * 4);
  float* cnt  = (float*)carve((size_t)G * 4);
  float* msq  = (float*)carve((size_t)G * 4);
  float* wsum = (float*)carve((size_t)G * 4);
  float* wbuf = (float*)carve((size_t)N * 4);
  float* bnsum = (float*)carve(256 * 4);
  float* bnss  = (float*)carve(256 * 4);
  float* bnsc  = (float*)carve(256 * 4);
  float* bnsh  = (float*)carve(256 * 4);
  (void)ws_size; (void)n_in;

  const int RPB = 256;                         // rows per block for channel kernels
  const int NB  = (N + RPB - 1) / RPB;

  const int* src = eidx;
  const int* dst = eidx + E;

  // graph node counts
  fill0(cnt, G, stream);
  count_nodes_kernel<<<(N + 255) / 256, 256, 0, stream>>>(batch, cnt, N);

  // edge embedding -> bf16
  edge_embed_kernel<<<E, 128, 0, stream>>>(edge_at, edge_w, edge_b, ebf, E);

  // node embedding: h = x @ node_w + node_b   (K=48 zero-padded)
  {
    dim3 grid((N + 63) / 64, HID / 64);
    gemm_wmma_kernel<<<grid, 256, 0, stream>>>(x, node_w, node_b, h, N, 48, HID, ACT_NONE);
  }

  for (int i = 0; i < L; ++i) {
    // z = (1+eps)*h
    scale_copy_kernel<<<4096, 256, 0, stream>>>(h, z, eps, i, (long)N * HID);

    // z += scatter(relu(h[src] + e @ cew[i] + ceb[i]))
    {
      dim3 grid((E + 63) / 64, HID / 64);
      edge_gemm_scatter_kernel<<<grid, 256, 0, stream>>>(
          ebf, cew + (size_t)i * 128 * HID, ceb + (size_t)i * HID,
          h, z, src, dst, E, 128, HID);
    }

    // t = relu(z @ mw1 + mb1); z = t @ mw2 + mb2
    {
      dim3 grid((N + 63) / 64, HID / 64);
      gemm_wmma_kernel<<<grid, 256, 0, stream>>>(
          z, mw1 + (size_t)i * HID * HID, mb1 + (size_t)i * HID, t, N, HID, HID, ACT_RELU);
      gemm_wmma_kernel<<<grid, 256, 0, stream>>>(
          t, mw2 + (size_t)i * HID * HID, mb2 + (size_t)i * HID, z, N, HID, HID, ACT_NONE);
    }

    // BatchNorm
    fill0(bnsum, 256, stream);
    fill0(bnss, 256, stream);
    bn_stats_kernel<<<NB, 256, 0, stream>>>(z, bnsum, bnss, N, RPB);
    bn_finalize_kernel<<<1, 256, 0, stream>>>(bnsum, bnss, bng + (size_t)i * 256,
                                              bnb + (size_t)i * 256, bnsc, bnsh, N);

    // PairNorm
    fill0(gsum, (long)G * HID, stream);
    fill0(msq, G, stream);
    bn_apply_gsum_kernel<<<NB, 256, 0, stream>>>(z, batch, gsum, bnsc, bnsh, N, RPB);
    pn_center_kernel<<<NB, 256, 0, stream>>>(z, batch, gsum, cnt, msq, N, RPB);
    pn_scale_relu_kernel<<<NB, 256, 0, stream>>>(z, batch, msq, cnt, h, N, RPB);
  }

  // pooling
  fill0(gsum, (long)G * HID, stream);
  fill0(maxb, (long)G * HID, stream);   // h >= 0, so 0.0f init matches reference
  fill0(atta, (long)G * HID, stream);
  fill0(wsum, G, stream);

  {
    dim3 grid((N + 63) / 64, 128 / 64);
    gemm_wmma_kernel<<<grid, 256, 0, stream>>>(h, att_w1, att_b1, t, N, HID, 128, ACT_TANH);
  }
  att_score_kernel<<<(N + 255) / 256, 256, 0, stream>>>(t, att_w2, att_b2, batch, wbuf, wsum, N);
  pool_accum_kernel<<<NB, 256, 0, stream>>>(h, batch, wbuf, wsum, gsum, maxb, atta, N, RPB);

  final_out_kernel<<<(int)(((long)G * 768 + 255) / 256), 256, 0, stream>>>(
      gsum, maxb, atta, cnt, out, G);
}
